// ConvTranBackbone_46273977647187
// MI455X (gfx1250) — matrix operand order repaired and checked by reading, not compiled, not run on hardware
//
#include <hip/hip_runtime.h>
#include <hip/hip_bf16.h>
#include <math.h>

// ---------------- types ----------------
typedef __attribute__((ext_vector_type(16))) _Float16 v16h;
typedef __attribute__((ext_vector_type(8)))  _Float16 v8h;
typedef __attribute__((ext_vector_type(8)))  float    v8f;
typedef __attribute__((ext_vector_type(4)))  unsigned int u32x4;
typedef __attribute__((ext_vector_type(8)))  int      i32x8;
typedef __attribute__((ext_vector_type(4)))  int      i32x4;

#define BB   16
#define CIN  32
#define SS   512
#define DD   256
#define HH   8
#define HD   32
#define LL   4
#define FF   1024
#define MM   (BB*SS)          // 8192 rows

#if defined(__has_builtin)
#if __has_builtin(__builtin_amdgcn_tensor_load_to_lds) && __has_builtin(__builtin_amdgcn_s_wait_tensorcnt)
#define USE_TDM 1
#endif
#endif
#ifndef USE_TDM
#define USE_TDM 0
#endif

__device__ __forceinline__ float gelu_f(float x) {
    return 0.5f * x * (1.0f + erff(x * 0.70710678118654752f));
}

// pack two 8x f16 chunks (16B each) into one v16h WMMA operand
__device__ __forceinline__ v16h ld16_2(const _Float16* lo, const _Float16* hi) {
    v8h l = *(const v8h*)lo;
    v8h h = *(const v8h*)hi;
    v16h r;
#pragma unroll
    for (int i = 0; i < 8; ++i) { r[i] = l[i]; r[i + 8] = h[i]; }
    return r;
}

// ---------------- weight transpose f32(R,C) -> f16(C,R) ----------------
__global__ void wtrans_k(const float* __restrict__ in, _Float16* __restrict__ out,
                         int R, int C) {
    int idx = blockIdx.x * 256 + threadIdx.x;
    if (idx >= R * C) return;
    int r = idx / C, c = idx - r * C;
    out[(size_t)c * R + r] = (_Float16)in[idx];
}

// ---------------- v transpose: (B*S, D) f16 -> (B,H,HD,S) f16 ----------------
__global__ void vtrans_k(const _Float16* __restrict__ vin, _Float16* __restrict__ vT) {
    int idx = blockIdx.x * 256 + threadIdx.x;      // over B*S*D
    int d = idx & (DD - 1);
    int s = (idx >> 8) & (SS - 1);
    int b = idx >> 17;
    int h = d >> 5, hd = d & (HD - 1);
    vT[((size_t)((b * HH + h) * HD + hd) << 9) + s] = vin[idx];
}

// ---------------- conv1 + bn + gelu : x(B,CIN,S) -> h1(B,D,S) ----------------
__global__ __launch_bounds__(256) void conv1_k(const float* __restrict__ x,
                                               const float* __restrict__ w,
                                               const float* __restrict__ cb,
                                               const float* __restrict__ bg,
                                               const float* __restrict__ bb,
                                               float* __restrict__ h1) {
    __shared__ float sx[CIN * 7];
    int t = blockIdx.x, b = blockIdx.y, o = threadIdx.x;
    if (threadIdx.x < CIN * 7) {
        int i = threadIdx.x / 7, k = threadIdx.x % 7;
        int tt = t + k - 3;
        sx[threadIdx.x] = (tt >= 0 && tt < SS) ? x[((size_t)b * CIN + i) * SS + tt] : 0.0f;
    }
    __syncthreads();
    float acc = cb[o];
    const float* wr = w + (size_t)o * (CIN * 7);
#pragma unroll 8
    for (int j = 0; j < CIN * 7; ++j) acc += wr[j] * sx[j];
    float y = acc * (bg[o] * rsqrtf(1.0f + 1e-5f)) + bb[o];
    h1[((size_t)b * DD + o) * SS + t] = gelu_f(y);
}

// ---------------- conv2 + bn + gelu + transpose + PE : h1 -> t(B,S,D) ----------------
__global__ __launch_bounds__(256) void conv2_k(const float* __restrict__ h1,
                                               const float* __restrict__ w,
                                               const float* __restrict__ cb,
                                               const float* __restrict__ bg,
                                               const float* __restrict__ bb,
                                               float* __restrict__ tout) {
    __shared__ float sh[DD * 36];                  // 36 KB
    int t0 = blockIdx.x * 32, b = blockIdx.y, o = threadIdx.x;
#pragma unroll
    for (int m = 0; m < 36; ++m) {
        int lidx = m * 256 + threadIdx.x;          // 0..9215
        int i = lidx / 36, tt = lidx % 36;
        int sg = t0 + tt - 2;
        sh[lidx] = (sg >= 0 && sg < SS) ? h1[((size_t)b * DD + i) * SS + sg] : 0.0f;
    }
    __syncthreads();
    float acc[32];
    float c0 = cb[o];
#pragma unroll
    for (int tt = 0; tt < 32; ++tt) acc[tt] = c0;
    const float* wr = w + (size_t)o * (DD * 5);
    for (int i = 0; i < DD; ++i) {
        const float* shl = sh + i * 36;
#pragma unroll
        for (int k = 0; k < 5; ++k) {
            float wv = wr[i * 5 + k];
#pragma unroll
            for (int tt = 0; tt < 32; ++tt) acc[tt] += wv * shl[tt + k];
        }
    }
    float bscale = bg[o] * rsqrtf(1.0f + 1e-5f);
    float bsh = bb[o];
    float e = (float)(o & ~1);
    float f = expf(e * (-9.210340371976184f / 256.0f));
#pragma unroll
    for (int tt = 0; tt < 32; ++tt) {
        int s = t0 + tt;
        float y = gelu_f(acc[tt] * bscale + bsh);
        float arg = (float)s * f * 0.5f;           // scale = D/S = 0.5
        y += (o & 1) ? cosf(arg) : sinf(arg);
        tout[((size_t)b * SS + s) * DD + o] = y;
    }
}

// ---------------- LayerNorm over D=256, one wave per row ----------------
template <bool F16OUT>
__global__ __launch_bounds__(256) void ln_k(const float* __restrict__ x,
                                            const float* __restrict__ g,
                                            const float* __restrict__ b,
                                            _Float16* __restrict__ outH,
                                            float* __restrict__ outF) {
    int wave = threadIdx.x >> 5, lane = threadIdx.x & 31;
    int row = blockIdx.x * 8 + wave;
    const float* xr = x + (size_t)row * DD;
    float v[8], s = 0.0f, s2 = 0.0f;
#pragma unroll
    for (int j = 0; j < 8; ++j) {
        v[j] = xr[lane + 32 * j];
        s += v[j]; s2 += v[j] * v[j];
    }
#pragma unroll
    for (int off = 16; off; off >>= 1) {
        s  += __shfl_xor(s,  off, 32);
        s2 += __shfl_xor(s2, off, 32);
    }
    float mu = s * (1.0f / DD);
    float var = s2 * (1.0f / DD) - mu * mu;
    float rstd = rsqrtf(var + 1e-5f);
#pragma unroll
    for (int j = 0; j < 8; ++j) {
        int col = lane + 32 * j;
        float y = (v[j] - mu) * rstd * g[col] + b[col];
        if (F16OUT) outH[(size_t)row * DD + col] = (_Float16)y;
        else        outF[(size_t)row * DD + col] = y;
    }
}

// ------------- wave GEMM: 32x64 per wave = 8 WMMA per 32-K step ------------------
// MODE 0: plain f16 store; MODE 1: += bias, residual-accumulate into f32; MODE 2: bias+gelu -> f16
template <int MODE>
__global__ __launch_bounds__(32) void gemm_wmma(const _Float16* __restrict__ A,
                                                const _Float16* __restrict__ Wt,
                                                const float* __restrict__ bias,
                                                float* __restrict__ outF,
                                                _Float16* __restrict__ outH,
                                                int M, int N, int K) {
    int lane = threadIdx.x;
    int lrow = lane & 15, hsel = lane >> 4;
    int kb8 = hsel * 8, kb16 = hsel * 16;
    int mbase = blockIdx.x * 32;
    int nbase = blockIdx.y * 64;
    const _Float16* arow0 = A + (size_t)(mbase + lrow) * K;
    const _Float16* arow1 = arow0 + (size_t)16 * K;

    v8f acc[2][4];
#pragma unroll
    for (int mt = 0; mt < 2; ++mt)
#pragma unroll
        for (int sN = 0; sN < 4; ++sN) acc[mt][sN] = {};

    for (int k0 = 0; k0 < K; k0 += 32) {
        // CDNA5 global_prefetch_b8 on next K chunk of the streamed A operand
        __builtin_prefetch(arow0 + k0 + 32, 0, 1);
        __builtin_prefetch(arow1 + k0 + 32, 0, 1);
        v16h a0 = ld16_2(arow0 + k0 + kb8, arow0 + k0 + kb8 + 16);
        v16h a1 = ld16_2(arow1 + k0 + kb8, arow1 + k0 + kb8 + 16);
#pragma unroll
        for (int sN = 0; sN < 4; ++sN) {
            const _Float16* bp = Wt + (size_t)(nbase + sN * 16 + lrow) * K + k0 + kb8;
            v16h bm = ld16_2(bp, bp + 16);
            acc[0][sN] = __builtin_amdgcn_wmma_f32_16x16x32_f16(
                false, a0, false, bm, (short)0, acc[0][sN], false, false);
            acc[1][sN] = __builtin_amdgcn_wmma_f32_16x16x32_f16(
                false, a1, false, bm, (short)0, acc[1][sN], false, false);
        }
    }
#pragma unroll
    for (int mt = 0; mt < 2; ++mt) {
#pragma unroll
        for (int sN = 0; sN < 4; ++sN) {
            int col = nbase + sN * 16 + lrow;
            float bv = (MODE == 0) ? 0.0f : bias[col];
#pragma unroll
            for (int g = 0; g < 8; ++g) {
                int row = mbase + mt * 16 + hsel * 8 + g;
                size_t idx = (size_t)row * N + col;
                float v = acc[mt][sN][g];
                if (MODE == 0)      outH[idx] = (_Float16)v;
                else if (MODE == 1) outF[idx] += v + bv;
                else                outH[idx] = (_Float16)gelu_f(v + bv);
            }
        }
    }
}

// ---------------- fused attention: one wave per (b, h, 16-row block) ----------------
// K-block (512 x 32 f16) staged into LDS by the Tensor Data Mover when available.
__global__ __launch_bounds__(32) void attn_k(const _Float16* __restrict__ q,
                                             const _Float16* __restrict__ k,
                                             const _Float16* __restrict__ vT,
                                             _Float16* __restrict__ ctx,
                                             const float* __restrict__ bt) {
    __shared__ float    sS[16 * SS];               // 32 KB scores
    __shared__ _Float16 sP[16 * SS];               // 16 KB probs
    __shared__ _Float16 sK[SS * HD];               // 32 KB K block (row j: 32 halfs)
    int lane = threadIdx.x;
    int lrow = lane & 15, hsel = lane >> 4;
    int kb8 = hsel * 8, kb16 = hsel * 16;
    int it = blockIdx.x, h = blockIdx.y, b = blockIdx.z;
    const float scale = 0.17677669529663687f;      // 1/sqrt(32)

    const _Float16* kblk = k + ((size_t)b * SS) * DD + h * HD;   // rows j, stride DD

#if USE_TDM
    {
        // Tensor DMA descriptor (D#), per CDNA5 ISA
        unsigned int lds_off = (unsigned int)(size_t)(void*)sK;  // LDS byte offset (addr[31:0])
        unsigned long long ga = (unsigned long long)(size_t)(const void*)kblk;
        u32x4 g0;
        g0[0] = 1u;                                  // count=1, is_restore=0, gather=0
        g0[1] = lds_off;                             // lds_addr
        g0[2] = (unsigned int)ga;                    // global_addr[31:0]
        g0[3] = (unsigned int)(ga >> 32) | (2u << 30); // global_addr[56:32] | type=2
        i32x8 g1;
        g1[0] = (int)(1u << 16);                     // data_size=2B code 1; no multicast/pad/iter
        g1[1] = (int)(32u << 16);                    // tensor_dim0 = 32 (bits 79:48 low part)
        g1[2] = (int)(512u << 16);                   // tensor_dim1 = 512 (bits 111:80 low part)
        g1[3] = (int)(32u << 16);                    // tile_dim0 = 32 (bits 127:112)
        g1[4] = 512;                                 // tile_dim1 = 512 (bits 143:128)
        g1[5] = 256;                                 // tensor_dim0_stride = 256 (bits 207:160)
        g1[6] = 0; g1[7] = 0;                        // tensor_dim1_stride = 0 (2D)
        i32x4 z4 = {0, 0, 0, 0};
#if __clang_major__ >= 23
        i32x8 z8 = {0, 0, 0, 0, 0, 0, 0, 0};
        __builtin_amdgcn_tensor_load_to_lds(g0, g1, z4, z4, z8, 0);
#else
        __builtin_amdgcn_tensor_load_to_lds(g0, g1, z4, z4, 0);
#endif
        __builtin_amdgcn_s_wait_tensorcnt(0);
    }
#else
    {
        // fallback: cooperative staging with plain loads
        for (int j = lane; j < SS; j += 32) {
            const v8h* src = (const v8h*)(kblk + (size_t)j * DD);
            v8h* dst = (v8h*)(sK + j * HD);
#pragma unroll
            for (int m = 0; m < 4; ++m) dst[m] = src[m];
        }
    }
#endif
    __syncthreads();

    // A operand: q tile (16 rows x K=32), kept in registers for whole pass
    const _Float16* qp = q + ((size_t)(b * SS + it * 16 + lrow)) * DD + h * HD;
    v16h a = ld16_2(qp + kb8, qp + kb8 + 16);

    for (int jt = 0; jt < SS / 16; ++jt) {
        const _Float16* kp = sK + (jt * 16 + lrow) * HD + kb8;
        v16h bm = ld16_2(kp, kp + 16);
        v8f c = {};
        c = __builtin_amdgcn_wmma_f32_16x16x32_f16(false, a, false, bm, (short)0, c, false, false);
        int j = jt * 16 + lrow;
#pragma unroll
        for (int g = 0; g < 8; ++g) {
            int r = hsel * 8 + g;
            int i = it * 16 + r;
            sS[r * SS + j] = c[g] * scale + bt[(size_t)(i - j + SS - 1) * HH + h];
        }
    }
    __syncthreads();

    // softmax along j (512), 16 rows, 32 lanes x 16 elems each
    for (int r = 0; r < 16; ++r) {
        float e[16], m = -3.4e38f;
#pragma unroll
        for (int jj = 0; jj < 16; ++jj) { e[jj] = sS[r * SS + lane + 32 * jj]; m = fmaxf(m, e[jj]); }
#pragma unroll
        for (int off = 16; off; off >>= 1) m = fmaxf(m, __shfl_xor(m, off, 32));
        float sum = 0.0f;
#pragma unroll
        for (int jj = 0; jj < 16; ++jj) { e[jj] = expf(e[jj] - m); sum += e[jj]; }
#pragma unroll
        for (int off = 16; off; off >>= 1) sum += __shfl_xor(sum, off, 32);
        float inv = 1.0f / sum;
#pragma unroll
        for (int jj = 0; jj < 16; ++jj) sP[r * SS + lane + 32 * jj] = (_Float16)(e[jj] * inv);
    }
    __syncthreads();

    // ctx(16 x 32) = P(16 x 512) @ V(512 x 32)
#pragma unroll
    for (int nt = 0; nt < 2; ++nt) {
        v8f cc = {};
        const _Float16* vp = vT + ((size_t)((b * HH + h) * HD + nt * 16 + lrow)) * SS;
        for (int kk = 0; kk < SS / 32; ++kk) {
            const _Float16* pp = &sP[lrow * SS + kk * 32 + kb8];
            v16h ap = ld16_2(pp, pp + 16);
            v16h bv = ld16_2(vp + kk * 32 + kb8, vp + kk * 32 + kb8 + 16);
            cc = __builtin_amdgcn_wmma_f32_16x16x32_f16(false, ap, false, bv, (short)0, cc, false, false);
        }
#pragma unroll
        for (int g = 0; g < 8; ++g) {
            int i = it * 16 + hsel * 8 + g;
            ctx[((size_t)(b * SS + i)) * DD + h * HD + nt * 16 + lrow] = (_Float16)cc[g];
        }
    }
}

// ---------------- launcher ----------------
extern "C" void kernel_launch(void* const* d_in, const int* in_sizes, int n_in,
                              void* d_out, int out_size, void* d_ws, size_t ws_size,
                              hipStream_t stream) {
    const float* x       = (const float*)d_in[0];
    const float* conv1_w = (const float*)d_in[1];
    const float* conv1_b = (const float*)d_in[2];
    const float* bn1_g   = (const float*)d_in[3];
    const float* bn1_b   = (const float*)d_in[4];
    const float* conv2_w = (const float*)d_in[5];
    const float* conv2_b = (const float*)d_in[6];
    const float* bn2_g   = (const float*)d_in[7];
    const float* bn2_b   = (const float*)d_in[8];
    const float* ln1_g   = (const float*)d_in[9];
    const float* ln1_b   = (const float*)d_in[10];
    const float* wq      = (const float*)d_in[11];
    const float* wk      = (const float*)d_in[12];
    const float* wv      = (const float*)d_in[13];
    const float* wo      = (const float*)d_in[14];
    const float* bo      = (const float*)d_in[15];
    const float* btab    = (const float*)d_in[16];
    const float* ln2_g   = (const float*)d_in[17];
    const float* ln2_b   = (const float*)d_in[18];
    const float* w1      = (const float*)d_in[19];
    const float* b1      = (const float*)d_in[20];
    const float* w2      = (const float*)d_in[21];
    const float* b2      = (const float*)d_in[22];
    const float* fn_g    = (const float*)d_in[23];
    const float* fn_b    = (const float*)d_in[24];

    // ---- workspace carve-up ----
    const size_t PER_L_HALFS = 4 * (size_t)DD * DD + 2 * (size_t)DD * FF; // 786432
    char* ws = (char*)d_ws;
    _Float16* wT   = (_Float16*)ws;                    ws += LL * PER_L_HALFS * 2;   // 6 MB
    float*    h1   = (float*)ws;                       ws += (size_t)BB*DD*SS*4;    // 8 MB
    float*    tbuf = (float*)ws;                       ws += (size_t)BB*SS*DD*4;    // 8 MB
    _Float16* xn   = (_Float16*)ws;                    ws += (size_t)MM*DD*2;       // 4 MB
    _Float16* qb   = (_Float16*)ws;                    ws += (size_t)MM*DD*2;
    _Float16* kb   = (_Float16*)ws;                    ws += (size_t)MM*DD*2;
    _Float16* vtmp = (_Float16*)ws;                    ws += (size_t)MM*DD*2;
    _Float16* vT   = (_Float16*)ws;                    ws += (size_t)MM*DD*2;
    _Float16* ctx  = (_Float16*)ws;                    ws += (size_t)MM*DD*2;
    _Float16* hff  = (_Float16*)ws;                    ws += (size_t)MM*FF*2;       // 16 MB
    (void)ws_size; (void)n_in; (void)in_sizes; (void)out_size;

    // ---- weight prep: f32 -> f16, transposed to (N,K) ----
    for (int l = 0; l < LL; ++l) {
        _Float16* base = wT + (size_t)l * PER_L_HALFS;
        _Float16* wqT = base;
        _Float16* wkT = wqT + (size_t)DD * DD;
        _Float16* wvT = wkT + (size_t)DD * DD;
        _Float16* woT = wvT + (size_t)DD * DD;
        _Float16* w1T = woT + (size_t)DD * DD;
        _Float16* w2T = w1T + (size_t)DD * FF;
        int nDD = DD * DD, nDF = DD * FF;
        wtrans_k<<<(nDD + 255) / 256, 256, 0, stream>>>(wq + (size_t)l * nDD, wqT, DD, DD);
        wtrans_k<<<(nDD + 255) / 256, 256, 0, stream>>>(wk + (size_t)l * nDD, wkT, DD, DD);
        wtrans_k<<<(nDD + 255) / 256, 256, 0, stream>>>(wv + (size_t)l * nDD, wvT, DD, DD);
        wtrans_k<<<(nDD + 255) / 256, 256, 0, stream>>>(wo + (size_t)l * nDD, woT, DD, DD);
        wtrans_k<<<(nDF + 255) / 256, 256, 0, stream>>>(w1 + (size_t)l * nDF, w1T, DD, FF);
        wtrans_k<<<(nDF + 255) / 256, 256, 0, stream>>>(w2 + (size_t)l * nDF, w2T, FF, DD);
    }

    // ---- tokenizer ----
    conv1_k<<<dim3(SS, BB), 256, 0, stream>>>(x, conv1_w, conv1_b, bn1_g, bn1_b, h1);
    conv2_k<<<dim3(SS / 32, BB), 256, 0, stream>>>(h1, conv2_w, conv2_b, bn2_g, bn2_b, tbuf);

    // ---- transformer layers ----
    for (int l = 0; l < LL; ++l) {
        _Float16* base = wT + (size_t)l * PER_L_HALFS;
        _Float16* wqT = base;
        _Float16* wkT = wqT + (size_t)DD * DD;
        _Float16* wvT = wkT + (size_t)DD * DD;
        _Float16* woT = wvT + (size_t)DD * DD;
        _Float16* w1T = woT + (size_t)DD * DD;
        _Float16* w2T = w1T + (size_t)DD * FF;

        ln_k<true><<<MM / 8, 256, 0, stream>>>(tbuf, ln1_g + l * DD, ln1_b + l * DD, xn, nullptr);

        gemm_wmma<0><<<dim3(MM / 32, DD / 64), 32, 0, stream>>>(xn, wqT, nullptr, nullptr, qb, MM, DD, DD);
        gemm_wmma<0><<<dim3(MM / 32, DD / 64), 32, 0, stream>>>(xn, wkT, nullptr, nullptr, kb, MM, DD, DD);
        gemm_wmma<0><<<dim3(MM / 32, DD / 64), 32, 0, stream>>>(xn, wvT, nullptr, nullptr, vtmp, MM, DD, DD);
        vtrans_k<<<(MM * DD) / 256, 256, 0, stream>>>(vtmp, vT);

        attn_k<<<dim3(SS / 16, HH, BB), 32, 0, stream>>>(qb, kb, vT, ctx,
                                                         btab + (size_t)l * (2 * SS - 1) * HH);

        gemm_wmma<1><<<dim3(MM / 32, DD / 64), 32, 0, stream>>>(ctx, woT, bo + l * DD, tbuf, nullptr, MM, DD, DD);

        ln_k<true><<<MM / 8, 256, 0, stream>>>(tbuf, ln2_g + l * DD, ln2_b + l * DD, xn, nullptr);

        gemm_wmma<2><<<dim3(MM / 32, FF / 64), 32, 0, stream>>>(xn, w1T, b1 + l * FF, nullptr, hff, MM, FF, DD);
        gemm_wmma<1><<<dim3(MM / 32, DD / 64), 32, 0, stream>>>(hff, w2T, b2 + l * DD, tbuf, nullptr, MM, DD, FF);
    }

    // ---- final LN -> f32 output ----
    ln_k<false><<<MM / 8, 256, 0, stream>>>(tbuf, fn_g, fn_b, nullptr, (float*)d_out);
}
